// Tree_TransformerLayer_55602646614246
// MI455X (gfx1250) — matrix operand order repaired and checked by reading, not compiled, not run on hardware
//
#include <hip/hip_runtime.h>
#include <hip/hip_bf16.h>

typedef __attribute__((ext_vector_type(16))) _Float16 v16h;
typedef __attribute__((ext_vector_type(8)))  _Float16 v8h;
typedef __attribute__((ext_vector_type(8)))  float    v8f;

#define DMODEL 256
#define DFF    1024
#define KIN    259        // d_model + 3
#define KPAD   288        // 9 k-tiles of 32
#define KT1    (KPAD/32)  // 9
#define KT2    (DFF/32)   // 32
#define NT1    (DFF/16)   // 64
#define NT2    (DMODEL/16)// 16
#define TM     64         // rows per block
#define MT     4          // m-tiles (16 rows each)
#define HC     128        // hidden chunk width
#define HCT    (HC/16)    // 8 hidden n-tiles per chunk (one per wave)
#define NCH    (DFF/HC)   // 8 chunks
#define XLD    296        // 592B rows: 16B aligned, conflict-free b128
#define HLD2   136        // 272B rows: 16B aligned, conflict-free b128
#define EPS    1e-5f

// ---------------- WMMA fragment helpers ----------------
// A-fragment (16x32 f16, M x K): lane L: m=L&15, half=L>>4;
// elems 0..7 -> K = ktile*32 + half*8 + 0..7, elems 8..15 -> +16.
__device__ __forceinline__ v16h load_a_frag(const _Float16* base, int ld,
                                            int ktile, int lane) {
    int m    = lane & 15;
    int half = lane >> 4;
    const _Float16* p = base + m * ld + ktile * 32 + half * 8;
    v8h lo = *(const v8h*)p;
    v8h hi = *(const v8h*)(p + 16);
    return __builtin_shufflevector(lo, hi, 0,1,2,3,4,5,6,7,
                                           8,9,10,11,12,13,14,15);
}

// B-fragment (32x16 f16): packed so each lane reads 16 contiguous f16 (32B).
__device__ __forceinline__ v16h load_b_frag(const _Float16* __restrict__ wp,
                                            int kt_count, int ntile, int ktile,
                                            int lane) {
    size_t idx = (((size_t)ntile * kt_count + ktile) * 32 + lane) * 16;
    return *(const v16h*)(wp + idx);
}

#define WMMA(a, b, c) __builtin_amdgcn_wmma_f32_16x16x32_f16( \
        false, (a), false, (b), (short)0, (c), false, false)

// ---------------- weight packing into B-fragment order ----------------
__global__ void pack_w_kernel(const float* __restrict__ W, _Float16* __restrict__ out,
                              int K, int Kpad, int Ncols) {
    int t = blockIdx.x * blockDim.x + threadIdx.x;
    int ktc = Kpad / 32;
    int total = ktc * (Ncols / 16) * 32;
    if (t >= total) return;
    int lane  = t & 31;
    int frag  = t >> 5;
    int ktile = frag % ktc;
    int ntile = frag / ktc;
    int n     = ntile * 16 + (lane & 15);
    int kbase = ktile * 32 + (lane >> 4) * 8;
    _Float16* dst = out + (size_t)t * 16;
    #pragma unroll
    for (int e = 0; e < 16; ++e) {
        int k = kbase + (e < 8 ? e : e + 8);
        float v = (k < K) ? W[(size_t)k * Ncols + n] : 0.0f;
        dst[e] = (_Float16)v;
    }
}

// ---------------- fused level kernel: 64 rows/block ----------------
__global__ __launch_bounds__(256)
void fused_level_kernel(const float* __restrict__ X,
                        const float* __restrict__ pcd_f,
                        const float* __restrict__ pcd_c,
                        const int*   __restrict__ inv, int N,
                        const _Float16* __restrict__ w3p,
                        const float* __restrict__ b3,
                        const _Float16* __restrict__ w4p,
                        const float* __restrict__ b4,
                        const float* __restrict__ gamma,
                        const float* __restrict__ beta,
                        float* __restrict__ sums,
                        float* __restrict__ cnts) {
    __shared__ __align__(16) _Float16 Xs[TM * XLD];   // 37,888 B
    __shared__ __align__(16) _Float16 Hs[TM * HLD2];  // 17,408 B
    __shared__ __align__(16) float    RS[TM][2];      //    512 B

    const int tid  = threadIdx.x;
    const int lane = tid & 31;
    const int wid  = tid >> 5;
    const int half = lane >> 4;
    const int lcol = lane & 15;
    const int r0   = blockIdx.x * TM;

    // ---- stage X tile (f32 -> f16), concat delta, zero pad ----
    for (int t = tid; t < TM * XLD; t += 256) {
        int row = t / XLD;
        int col = t - row * XLD;
        int r   = r0 + row;
        float v = 0.0f;
        if (r < N) {
            if (col < DMODEL) {
                v = X[(size_t)r * DMODEL + col];
            } else if (col < DMODEL + 3) {
                int c3 = col - DMODEL;
                int s  = inv[r];
                v = pcd_f[(size_t)r * 3 + c3] - pcd_c[(size_t)s * 3 + c3];
            }
        }
        Xs[t] = (_Float16)v;
    }

    // persistent GEMM2 accumulators: 4 m-tiles x 2 n-tiles per wave
    v8f acc2[MT][2];
    #pragma unroll
    for (int m = 0; m < MT; ++m) { acc2[m][0] = (v8f){}; acc2[m][1] = (v8f){}; }

    const int n0 = 2 * wid;  // this wave's output n-tiles: n0, n0+1

    // ---- chunked hidden: GEMM1 -> LDS chunk -> GEMM2 accumulate ----
    for (int c = 0; c < NCH; ++c) {
        __syncthreads();   // X staged / previous chunk consumed

        // GEMM1: this wave computes hidden n-tile g for all 4 m-tiles
        const int g = c * HCT + wid;
        v8f acc1[MT];
        #pragma unroll
        for (int m = 0; m < MT; ++m) acc1[m] = (v8f){};
        #pragma unroll
        for (int kt = 0; kt < KT1; ++kt) {
            v16h b = load_b_frag(w3p, KT1, g, kt, lane);
            #pragma unroll
            for (int m = 0; m < MT; ++m) {
                v16h a = load_a_frag(Xs + m * 16 * XLD, XLD, kt, lane);
                acc1[m] = WMMA(a, b, acc1[m]);
            }
        }
        {
            float bias = b3[g * 16 + lcol];
            #pragma unroll
            for (int m = 0; m < MT; ++m) {
                #pragma unroll
                for (int v = 0; v < 8; ++v) {
                    float h = acc1[m][v] + bias;
                    h = h > 0.0f ? h : 0.0f;
                    Hs[(m * 16 + v + 8 * half) * HLD2 + wid * 16 + lcol] =
                        (_Float16)h;
                }
            }
        }
        __syncthreads();   // hidden chunk ready

        // GEMM2 partial accumulate over this chunk (K = 128 -> 4 k-tiles)
        #pragma unroll
        for (int kt = 0; kt < HC / 32; ++kt) {
            int ktg = c * (HC / 32) + kt;
            v16h b0 = load_b_frag(w4p, KT2, n0,     ktg, lane);
            v16h b1 = load_b_frag(w4p, KT2, n0 + 1, ktg, lane);
            #pragma unroll
            for (int m = 0; m < MT; ++m) {
                v16h a = load_a_frag(Hs + m * 16 * HLD2, HLD2, kt, lane);
                acc2[m][0] = WMMA(a, b0, acc2[m][0]);
                acc2[m][1] = WMMA(a, b1, acc2[m][1]);
            }
        }
    }

    // ---- epilogue: bias, LayerNorm (register-resident), scatter ----
    __syncthreads();
    if (tid < TM * 2) ((float*)RS)[tid] = 0.0f;
    // per-row count atomics (once per row)
    if (tid < TM) {
        int r = r0 + tid;
        if (r < N) atomicAdd(cnts + inv[r], 1.0f);
    }
    __syncthreads();

    const int col0 = n0 * 16 + lcol;
    const int col1 = (n0 + 1) * 16 + lcol;
    const float bias0 = b4[col0];
    const float bias1 = b4[col1];

    #pragma unroll
    for (int m = 0; m < MT; ++m) {
        #pragma unroll
        for (int v = 0; v < 8; ++v) {
            float a0 = acc2[m][0][v] + bias0;
            float a1 = acc2[m][1][v] + bias1;
            acc2[m][0][v] = a0;
            acc2[m][1][v] = a1;
            float s  = a0 + a1;
            float s2 = a0 * a0 + a1 * a1;
            #pragma unroll
            for (int o = 8; o > 0; o >>= 1) {  // reduce across the 16-lane half
                s  += __shfl_xor(s,  o, 32);
                s2 += __shfl_xor(s2, o, 32);
            }
            if (lcol == 0) {
                int row = m * 16 + v + 8 * half;
                atomicAdd(&RS[row][0], s);
                atomicAdd(&RS[row][1], s2);
            }
        }
    }
    __syncthreads();

    const float g0 = gamma[col0], be0 = beta[col0];
    const float g1 = gamma[col1], be1 = beta[col1];
    #pragma unroll
    for (int m = 0; m < MT; ++m) {
        #pragma unroll
        for (int v = 0; v < 8; ++v) {
            int row = m * 16 + v + 8 * half;
            int r   = r0 + row;
            if (r < N) {
                float mu  = RS[row][0] * (1.0f / DMODEL);
                float var = RS[row][1] * (1.0f / DMODEL) - mu * mu;
                float rs  = rsqrtf(var + EPS);
                int seg   = inv[r];
                float* dst = sums + (size_t)seg * DMODEL;
                atomicAdd(dst + col0, (acc2[m][0][v] - mu) * rs * g0 + be0);
                atomicAdd(dst + col1, (acc2[m][1][v] - mu) * rs * g1 + be1);
            }
        }
    }
}

// ---------------- finalize: mean = sum / max(count,1), in place ----------------
__global__ void finalize_kernel(float* __restrict__ data,
                                const float* __restrict__ cnts, int nseg) {
    int t = blockIdx.x * blockDim.x + threadIdx.x;
    if (t >= nseg * DMODEL) return;
    int seg = t >> 8;
    data[t] = data[t] / fmaxf(cnts[seg], 1.0f);
}

extern "C" void kernel_launch(void* const* d_in, const int* in_sizes, int n_in,
                              void* d_out, int out_size, void* d_ws, size_t ws_size,
                              hipStream_t stream) {
    const float* feature = (const float*)d_in[0];
    const float* pcd0    = (const float*)d_in[1];
    const float* pcd1    = (const float*)d_in[2];
    const float* pcd2    = (const float*)d_in[3];
    const float* W3      = (const float*)d_in[4];
    const float* b3      = (const float*)d_in[5];
    const float* W4      = (const float*)d_in[6];
    const float* b4      = (const float*)d_in[7];
    const float* gamma   = (const float*)d_in[8];
    const float* beta    = (const float*)d_in[9];
    const int*   inv0    = (const int*)d_in[10];
    const int*   inv1    = (const int*)d_in[11];

    const int N0 = in_sizes[0] / DMODEL;  // 100000
    const int N1 = in_sizes[2] / 3;       // 25000
    const int N2 = in_sizes[3] / 3;       // 6250

    // workspace: packed weights + counts
    char* ws = (char*)d_ws;
    size_t o = 0;
    auto take = [&](size_t bytes) { char* p = ws + o; o = (o + bytes + 255) & ~(size_t)255; return p; };
    const size_t w3p_bytes = (size_t)KT1 * NT1 * 512 * sizeof(_Float16);
    const size_t w4p_bytes = (size_t)KT2 * NT2 * 512 * sizeof(_Float16);
    _Float16* w3p = (_Float16*)take(w3p_bytes);
    _Float16* w4p = (_Float16*)take(w4p_bytes);
    float*    cnt0 = (float*)take((size_t)N1 * sizeof(float));
    float*    cnt1 = (float*)take((size_t)N2 * sizeof(float));

    float* out = (float*)d_out;
    float* f1  = out + (size_t)N0 * DMODEL;
    float* f2  = out + (size_t)(N0 + N1) * DMODEL;

    // pack weights (deterministic, every call)
    {
        int t3 = KT1 * NT1 * 32;
        pack_w_kernel<<<(t3 + 255) / 256, 256, 0, stream>>>(W3, w3p, KIN, KPAD, DFF);
        int t4 = KT2 * NT2 * 32;
        pack_w_kernel<<<(t4 + 255) / 256, 256, 0, stream>>>(W4, w4p, DFF, DFF, DMODEL);
    }

    // passthrough: out[0:N0] = feature
    hipMemcpyAsync(out, feature, (size_t)N0 * DMODEL * sizeof(float),
                   hipMemcpyDeviceToDevice, stream);

    // ---- level 0: feature -> f1 ----
    hipMemsetAsync(f1,   0, (size_t)N1 * DMODEL * sizeof(float), stream);
    hipMemsetAsync(cnt0, 0, (size_t)N1 * sizeof(float), stream);
    fused_level_kernel<<<(N0 + TM - 1) / TM, 256, 0, stream>>>(
        feature, pcd0, pcd1, inv0, N0, w3p, b3, w4p, b4, gamma, beta, f1, cnt0);
    finalize_kernel<<<((size_t)N1 * DMODEL + 255) / 256, 256, 0, stream>>>(f1, cnt0, N1);

    // ---- level 1: f1 -> f2 ----
    hipMemsetAsync(f2,   0, (size_t)N2 * DMODEL * sizeof(float), stream);
    hipMemsetAsync(cnt1, 0, (size_t)N2 * sizeof(float), stream);
    fused_level_kernel<<<(N1 + TM - 1) / TM, 256, 0, stream>>>(
        f1, pcd1, pcd2, inv1, N1, w3p, b3, w4p, b4, gamma, beta, f2, cnt1);
    finalize_kernel<<<((size_t)N2 * DMODEL + 255) / 256, 256, 0, stream>>>(f2, cnt1, N2);

    (void)n_in; (void)out_size; (void)ws_size; (void)N2;
}
